// Decoder_48833778155748
// MI455X (gfx1250) — compile-verified, matrix operand order checked by reading, not run on hardware
//
#include <hip/hip_runtime.h>
#include <hip/hip_bf16.h>
#include <math.h>

// Problem constants
#define B_   1024
#define Z_   256
#define TD_  512
#define CF_  1024
#define A_   64
#define R_   128
#define T_   64

typedef __attribute__((ext_vector_type(16))) __bf16 v16bf;
typedef __attribute__((ext_vector_type(8)))  float  v8f;

__device__ __forceinline__ unsigned short f32_to_bf16_rne(float f) {
    unsigned int u = __float_as_uint(f);
    unsigned int r = u + 0x7FFFu + ((u >> 16) & 1u);
    return (unsigned short)(r >> 16);
}

// ---------------------------------------------------------------------------
// Generic bf16 WMMA GEMM: Out[M,N] = A[M,K](bf16) @ B[N,K](bf16)^T (+bias)(relu)
// Block tile 128x128, 256 threads (8 waves), wave tile 32x64 (2x4 WMMA tiles).
// K staged 64-deep in LDS via global_load_async_to_lds_b128 (ASYNCcnt path).
// ---------------------------------------------------------------------------
#define TM 128
#define TN 128
#define KB 64
#define LDST 72   // 64 halves + 8 pad: 144B row stride -> conflict-free b128 reads

enum { GF_BIAS = 1, GF_RELU = 2, GF_OUTBF16 = 4 };

__device__ __forceinline__ void async_b128_x2(unsigned lds_byte_addr,
                                              const unsigned short* g) {
    // Copies 32 bytes (2 x b128); IOFFSET advances BOTH global and LDS address
    // (ISA 08_async_tensor: the LDS address also adds the instruction offset).
    asm volatile("global_load_async_to_lds_b128 %0, %1, off"
                 :: "v"(lds_byte_addr), "v"(g) : "memory");
    asm volatile("global_load_async_to_lds_b128 %0, %1, off offset:16"
                 :: "v"(lds_byte_addr), "v"(g) : "memory");
}

__global__ __launch_bounds__(256)
void wmma_gemm(const unsigned short* __restrict__ Amat, int lda,
               const unsigned short* __restrict__ Bmat, int ldb,
               void* __restrict__ Out, int ldc,
               const float* __restrict__ bias,
               int N, int K, int flags)
{
    __shared__ __align__(16) unsigned short As[TM * LDST];
    __shared__ __align__(16) unsigned short Bs[TN * LDST];

    const int tid  = threadIdx.x;
    const int m0   = blockIdx.y * TM;
    const int n0   = blockIdx.x * TN;
    const int wave = tid >> 5;
    const int lane = tid & 31;
    const int wm   = wave >> 1;           // 0..3 -> 32-row slab
    const int wn   = wave & 1;            // 0..1 -> 64-col slab
    const int lr   = lane & 15;
    const int lk   = (lane >> 4) << 4;    // 0 or 16 (K half within a k-step)

    const int lrow   = tid >> 1;          // 0..127 tile row for staging loads
    const int lquart = (tid & 1) << 5;    // 0 or 32 halves (64B) within the row

    const bool fullN = (n0 + TN) <= N;    // uniform per block: hot GEMMs always true

    const unsigned ldsA = (unsigned)(uintptr_t)&As[lrow * LDST + lquart];
    const unsigned ldsB = (unsigned)(uintptr_t)&Bs[lrow * LDST + lquart];

    v8f acc[2][4] = {};

    for (int k0 = 0; k0 < K; k0 += KB) {
        // ---- stage A tile (128 x 64 bf16) via async copy to LDS
        async_b128_x2(ldsA, Amat + (size_t)(m0 + lrow) * lda + k0 + lquart);

        // ---- stage B tile (128 x 64 bf16)
        if (fullN) {
            async_b128_x2(ldsB, Bmat + (size_t)(n0 + lrow) * ldb + k0 + lquart);
        } else {
            uint4* dst = (uint4*)(Bs + lrow * LDST + lquart);
            if (n0 + lrow < N) {
                const uint4* src = (const uint4*)(Bmat + (size_t)(n0 + lrow) * ldb + k0 + lquart);
                dst[0] = src[0]; dst[1] = src[1];
            } else {
                uint4 zz; zz.x = zz.y = zz.z = zz.w = 0u;
                dst[0] = zz; dst[1] = zz;
            }
        }
        asm volatile("s_wait_asynccnt 0x0" ::: "memory");
        __syncthreads();

        union Frag { v16bf v; uint4 u[2]; };
        #pragma unroll
        for (int ks = 0; ks < 2; ++ks) {           // two K=32 WMMA steps per stage
            Frag fa[2], fb[4];
            #pragma unroll
            for (int mi = 0; mi < 2; ++mi) {
                const uint4* p = (const uint4*)(As + (wm * 32 + mi * 16 + lr) * LDST + ks * 32 + lk);
                fa[mi].u[0] = p[0]; fa[mi].u[1] = p[1];
            }
            #pragma unroll
            for (int nj = 0; nj < 4; ++nj) {
                const uint4* p = (const uint4*)(Bs + (wn * 64 + nj * 16 + lr) * LDST + ks * 32 + lk);
                fb[nj].u[0] = p[0]; fb[nj].u[1] = p[1];
            }
            #pragma unroll
            for (int mi = 0; mi < 2; ++mi)
                #pragma unroll
                for (int nj = 0; nj < 4; ++nj)
                    acc[mi][nj] = __builtin_amdgcn_wmma_f32_16x16x32_bf16(
                        false, fa[mi].v, false, fb[nj].v,
                        (short)0, acc[mi][nj], false, false);
        }
        __syncthreads();
    }

    // Epilogue. 16x16 f32 C layout: VGPR r -> M = r + 8*(lane>=16), N = lane&15.
    #pragma unroll
    for (int mi = 0; mi < 2; ++mi) {
        #pragma unroll
        for (int nj = 0; nj < 4; ++nj) {
            const int col = n0 + wn * 64 + nj * 16 + lr;
            if (col >= N) continue;
            const float bv = (flags & GF_BIAS) ? bias[col] : 0.0f;
            const int rbase = m0 + wm * 32 + mi * 16 + ((lane >> 4) << 3);
            #pragma unroll
            for (int r = 0; r < 8; ++r) {
                float v = acc[mi][nj][r] + bv;
                if (flags & GF_RELU) v = fmaxf(v, 0.0f);
                const size_t o = (size_t)(rbase + r) * ldc + col;
                if (flags & GF_OUTBF16) ((unsigned short*)Out)[o] = f32_to_bf16_rne(v);
                else                    ((float*)Out)[o] = v;
            }
        }
    }
}

// ---------------------------------------------------------------------------
// f32 -> bf16 conversion with row re-stride (used to slice Wih[:, :512] too)
// ---------------------------------------------------------------------------
__global__ void cvt_f32_to_bf16(const float* __restrict__ src, int ld_src,
                                unsigned short* __restrict__ dst, int ld_dst,
                                int rows, int cols)
{
    int i = blockIdx.x * blockDim.x + threadIdx.x;
    if (i >= rows * cols) return;
    int r = i / cols, c = i - r * cols;
    dst[(size_t)r * ld_dst + c] = f32_to_bf16_rne(src[(size_t)r * ld_src + c]);
}

// ---------------------------------------------------------------------------
// Fused LSTM pointwise update.
// gate(b,jj) = gates[b,jj] + gpre[b,jj] + Wih[jj,c1] (+Wih[jj,c2]) (+ts*Wih[jj,cts])
// c' = sig(f)*c + sig(i)*tanh(g);  h = sig(o)*tanh(c')  (h written as bf16)
// ---------------------------------------------------------------------------
__global__ __launch_bounds__(256)
void lstm_update(const float* __restrict__ gates, const float* __restrict__ gpre,
                 const float* __restrict__ Wih, int ldWih,
                 const int* __restrict__ idx1, int colbase1,
                 const int* __restrict__ idx2, int colbase2,
                 const float* __restrict__ tsv, int col_ts,
                 float* __restrict__ c, unsigned short* __restrict__ h)
{
    const int b = blockIdx.y;
    const int j = blockIdx.x * blockDim.x + threadIdx.x;
    const int c1 = colbase1 + idx1[b];
    const int c2 = idx2 ? (colbase2 + idx2[b]) : 0;
    const float tv = tsv ? tsv[b] : 0.0f;

    float g4[4];
    #pragma unroll
    for (int k = 0; k < 4; ++k) {
        const int jj = k * CF_ + j;
        float v = gates[(size_t)b * 4 * CF_ + jj] + gpre[(size_t)b * 4 * CF_ + jj];
        v += Wih[(size_t)jj * ldWih + c1];
        if (idx2) v += Wih[(size_t)jj * ldWih + c2];
        if (tsv)  v += tv * Wih[(size_t)jj * ldWih + col_ts];
        g4[k] = v;
    }
    const float ig = 1.0f / (1.0f + expf(-g4[0]));
    const float fg = 1.0f / (1.0f + expf(-g4[1]));
    const float gg = tanhf(g4[2]);
    const float og = 1.0f / (1.0f + expf(-g4[3]));
    const size_t o = (size_t)b * CF_ + j;
    const float cn = fg * c[o] + ig * gg;
    c[o] = cn;
    h[o] = f32_to_bf16_rne(og * tanhf(cn));
}

// ---------------------------------------------------------------------------
__global__ void argmax_rows(const float* __restrict__ x, int ld, int N,
                            int* __restrict__ idx)
{
    const int b = blockIdx.x * blockDim.x + threadIdx.x;
    if (b >= B_) return;
    const float* row = x + (size_t)b * ld;
    float best = row[0]; int bi = 0;
    for (int n = 1; n < N; ++n) { float v = row[n]; if (v > best) { best = v; bi = n; } }
    idx[b] = bi;
}

// ts head second layer: ts[b] = dot(hidden[b,:512], w2) + b2  (wave-per-row)
__global__ __launch_bounds__(256)
void ts_head(const float* __restrict__ hidden, const float* __restrict__ w2,
             const float* __restrict__ b2, float* __restrict__ tsv,
             float* __restrict__ ts_out, int t)
{
    const int wave = threadIdx.x >> 5, lane = threadIdx.x & 31;
    const int b = blockIdx.x * 8 + wave;
    float s = 0.0f;
    for (int k = lane; k < TD_; k += 32) s += hidden[(size_t)b * TD_ + k] * w2[k];
    for (int off = 16; off; off >>= 1) s += __shfl_xor(s, off, 32);
    if (lane == 0) {
        const float v = s + b2[0];
        tsv[b] = v;
        ts_out[(size_t)b * T_ + t] = v;
    }
}

// in-place row softmax (wave-per-row, N = 64 or 128)
__global__ __launch_bounds__(256)
void softmax_rows(float* __restrict__ x, int N)
{
    const int wave = threadIdx.x >> 5, lane = threadIdx.x & 31;
    float* p = x + ((size_t)blockIdx.x * 8 + wave) * N;
    float m = -3.4e38f;
    for (int n = lane; n < N; n += 32) m = fmaxf(m, p[n]);
    for (int off = 16; off; off >>= 1) m = fmaxf(m, __shfl_xor(m, off, 32));
    float s = 0.0f;
    for (int n = lane; n < N; n += 32) { float e = expf(p[n] - m); p[n] = e; s += e; }
    for (int off = 16; off; off >>= 1) s += __shfl_xor(s, off, 32);
    const float inv = 1.0f / s;
    for (int n = lane; n < N; n += 32) p[n] *= inv;
}

__global__ void init_idx(int* __restrict__ aidx, int* __restrict__ ridx)
{
    const int b = blockIdx.x * blockDim.x + threadIdx.x;
    if (b < B_) { aidx[b] = A_ - 1; ridx[b] = R_ - 1; }
}

// ---------------------------------------------------------------------------
extern "C" void kernel_launch(void* const* d_in, const int* in_sizes, int n_in,
                              void* d_out, int out_size, void* d_ws, size_t ws_size,
                              hipStream_t stream)
{
    const float* z       = (const float*)d_in[0];
    const float* z2t_w   = (const float*)d_in[1];
    const float* z2t_b   = (const float*)d_in[2];
    const float* Wih_a   = (const float*)d_in[3];
    const float* Whh_a   = (const float*)d_in[4];
    const float* b_a     = (const float*)d_in[5];
    const float* Wih_r   = (const float*)d_in[6];
    const float* Whh_r   = (const float*)d_in[7];
    const float* b_r     = (const float*)d_in[8];
    const float* Wih_t   = (const float*)d_in[9];
    const float* Whh_t   = (const float*)d_in[10];
    const float* b_t     = (const float*)d_in[11];
    const float* e2act_w = (const float*)d_in[12];
    const float* e2act_b = (const float*)d_in[13];
    const float* e2res_w = (const float*)d_in[14];
    const float* e2res_b = (const float*)d_in[15];
    const float* e2ts_w1 = (const float*)d_in[16];
    const float* e2ts_b1 = (const float*)d_in[17];
    const float* e2ts_w2 = (const float*)d_in[18];
    const float* e2ts_b2 = (const float*)d_in[19];

    float* acts_out = (float*)d_out;                     // [B, T, A]
    float* ts_out   = acts_out + (size_t)B_ * T_ * A_;   // [B, T]
    float* res_out  = ts_out + (size_t)B_ * T_;          // [B, T, R]

    char* ws = (char*)d_ws;
    size_t off = 0;
    auto alloc = [&](size_t bytes) -> char* {
        char* p = ws + off;
        off += (bytes + 255) & ~(size_t)255;
        return p;
    };
    unsigned short* z_bf    = (unsigned short*)alloc((size_t)B_ * Z_ * 2);
    unsigned short* z2t_bf  = (unsigned short*)alloc((size_t)TD_ * Z_ * 2);
    unsigned short* trec_bf = (unsigned short*)alloc((size_t)B_ * TD_ * 2);
    unsigned short* wihA_bf = (unsigned short*)alloc((size_t)4 * CF_ * TD_ * 2);
    unsigned short* wihR_bf = (unsigned short*)alloc((size_t)4 * CF_ * TD_ * 2);
    unsigned short* wihT_bf = (unsigned short*)alloc((size_t)4 * CF_ * TD_ * 2);
    unsigned short* whhA_bf = (unsigned short*)alloc((size_t)4 * CF_ * CF_ * 2);
    unsigned short* whhR_bf = (unsigned short*)alloc((size_t)4 * CF_ * CF_ * 2);
    unsigned short* whhT_bf = (unsigned short*)alloc((size_t)4 * CF_ * CF_ * 2);
    unsigned short* eact_bf = (unsigned short*)alloc((size_t)A_ * CF_ * 2);
    unsigned short* eres_bf = (unsigned short*)alloc((size_t)R_ * CF_ * 2);
    unsigned short* ets1_bf = (unsigned short*)alloc((size_t)TD_ * CF_ * 2);
    float* gpreA  = (float*)alloc((size_t)B_ * 4 * CF_ * 4);
    float* gpreR  = (float*)alloc((size_t)B_ * 4 * CF_ * 4);
    float* gpreT  = (float*)alloc((size_t)B_ * 4 * CF_ * 4);
    float* gates  = (float*)alloc((size_t)B_ * 4 * CF_ * 4);
    unsigned short* hA_bf = (unsigned short*)alloc((size_t)B_ * CF_ * 2);
    unsigned short* hR_bf = (unsigned short*)alloc((size_t)B_ * CF_ * 2);
    unsigned short* hT_bf = (unsigned short*)alloc((size_t)B_ * CF_ * 2);
    float* cA     = (float*)alloc((size_t)B_ * CF_ * 4);
    float* cR     = (float*)alloc((size_t)B_ * CF_ * 4);
    float* cT     = (float*)alloc((size_t)B_ * CF_ * 4);
    float* hidden = (float*)alloc((size_t)B_ * TD_ * 4);
    float* tsv    = (float*)alloc((size_t)B_ * 4);
    int*   aidx   = (int*)alloc((size_t)B_ * 4);
    int*   ridx   = (int*)alloc((size_t)B_ * 4);

    auto cvt = [&](const float* src, int lds_, unsigned short* dst, int ldd,
                   int rows, int cols) {
        int total = rows * cols;
        cvt_f32_to_bf16<<<(total + 255) / 256, 256, 0, stream>>>(src, lds_, dst, ldd, rows, cols);
    };
    auto gemm = [&](const unsigned short* A, int lda, const unsigned short* Bw, int ldb,
                    void* Out, int ldc, const float* bias, int N, int K, int flags) {
        dim3 grid((N + TN - 1) / TN, B_ / TM);
        wmma_gemm<<<grid, 256, 0, stream>>>(A, lda, Bw, ldb, Out, ldc, bias, N, K, flags);
    };

    // --- weight/activation conversions to bf16 (once per call, deterministic)
    cvt(z,       Z_,            z_bf,    Z_,  B_,      Z_);
    cvt(z2t_w,   Z_,            z2t_bf,  Z_,  TD_,     Z_);
    cvt(Wih_a,   TD_ + A_,      wihA_bf, TD_, 4 * CF_, TD_);   // only t_rec part
    cvt(Wih_r,   TD_ + A_ + R_, wihR_bf, TD_, 4 * CF_, TD_);
    cvt(Wih_t,   TD_ + A_ + 1,  wihT_bf, TD_, 4 * CF_, TD_);
    cvt(Whh_a,   CF_,           whhA_bf, CF_, 4 * CF_, CF_);
    cvt(Whh_r,   CF_,           whhR_bf, CF_, 4 * CF_, CF_);
    cvt(Whh_t,   CF_,           whhT_bf, CF_, 4 * CF_, CF_);
    cvt(e2act_w, CF_,           eact_bf, CF_, A_,      CF_);
    cvt(e2res_w, CF_,           eres_bf, CF_, R_,      CF_);
    cvt(e2ts_w1, CF_,           ets1_bf, CF_, TD_,     CF_);

    // --- carry init
    hipMemsetAsync(hA_bf, 0, (size_t)B_ * CF_ * 2, stream);
    hipMemsetAsync(hR_bf, 0, (size_t)B_ * CF_ * 2, stream);
    hipMemsetAsync(hT_bf, 0, (size_t)B_ * CF_ * 2, stream);
    hipMemsetAsync(cA, 0, (size_t)B_ * CF_ * 4, stream);
    hipMemsetAsync(cR, 0, (size_t)B_ * CF_ * 4, stream);
    hipMemsetAsync(cT, 0, (size_t)B_ * CF_ * 4, stream);
    hipMemsetAsync(tsv, 0, (size_t)B_ * 4, stream);
    init_idx<<<4, 256, 0, stream>>>(aidx, ridx);

    // --- t_rec = relu(z @ z2t_w^T + b)  -> bf16
    gemm(z_bf, Z_, z2t_bf, Z_, trec_bf, TD_, z2t_b, TD_, Z_, GF_BIAS | GF_RELU | GF_OUTBF16);

    // --- loop-invariant input projections: g_pre = t_rec @ Wih[:, :512]^T + b
    gemm(trec_bf, TD_, wihA_bf, TD_, gpreA, 4 * CF_, b_a, 4 * CF_, TD_, GF_BIAS);
    gemm(trec_bf, TD_, wihR_bf, TD_, gpreR, 4 * CF_, b_r, 4 * CF_, TD_, GF_BIAS);
    gemm(trec_bf, TD_, wihT_bf, TD_, gpreT, 4 * CF_, b_t, 4 * CF_, TD_, GF_BIAS);

    const dim3 lgrid(CF_ / 256, B_);
    for (int t = 0; t < T_; ++t) {
        // cell a (uses previous act one-hot)
        gemm(hA_bf, CF_, whhA_bf, CF_, gates, 4 * CF_, nullptr, 4 * CF_, CF_, 0);
        lstm_update<<<lgrid, 256, 0, stream>>>(gates, gpreA, Wih_a, TD_ + A_,
                                               aidx, TD_, nullptr, 0, nullptr, 0, cA, hA_bf);
        gemm(hA_bf, CF_, eact_bf, CF_, acts_out + (size_t)t * A_, T_ * A_,
             e2act_b, A_, CF_, GF_BIAS);
        argmax_rows<<<4, 256, 0, stream>>>(acts_out + (size_t)t * A_, T_ * A_, A_, aidx);

        // cell r (uses new act one-hot, previous res one-hot)
        gemm(hR_bf, CF_, whhR_bf, CF_, gates, 4 * CF_, nullptr, 4 * CF_, CF_, 0);
        lstm_update<<<lgrid, 256, 0, stream>>>(gates, gpreR, Wih_r, TD_ + A_ + R_,
                                               aidx, TD_, ridx, TD_ + A_, nullptr, 0, cR, hR_bf);
        gemm(hR_bf, CF_, eres_bf, CF_, res_out + (size_t)t * R_, T_ * R_,
             e2res_b, R_, CF_, GF_BIAS);
        argmax_rows<<<4, 256, 0, stream>>>(res_out + (size_t)t * R_, T_ * R_, R_, ridx);

        // cell t (uses new act one-hot, previous ts scalar)
        gemm(hT_bf, CF_, whhT_bf, CF_, gates, 4 * CF_, nullptr, 4 * CF_, CF_, 0);
        lstm_update<<<lgrid, 256, 0, stream>>>(gates, gpreT, Wih_t, TD_ + A_ + 1,
                                               aidx, TD_, nullptr, 0, tsv, TD_ + A_, cT, hT_bf);
        gemm(hT_bf, CF_, ets1_bf, CF_, hidden, TD_, e2ts_b1, TD_, CF_, GF_BIAS | GF_RELU);
        ts_head<<<B_ / 8, 256, 0, stream>>>(hidden, e2ts_w2, e2ts_b2, tsv, ts_out, t);
    }

    // --- final softmaxes (in place on d_out)
    softmax_rows<<<(B_ * T_) / 8, 256, 0, stream>>>(acts_out, A_);
    softmax_rows<<<(B_ * T_) / 8, 256, 0, stream>>>(res_out, R_);
}
